// NetVLAD_26620207301332
// MI455X (gfx1250) — matrix operand order repaired and checked by reading, not compiled, not run on hardware
//
#include <hip/hip_runtime.h>
#include <hip/hip_bf16.h>

// NetVLAD forward for MI455X (gfx1250, wave32, WMMA).
// Pipeline: pack/transpose+norms -> GEMM1(WMMA f16)+BN+LIF+softmax -> asum
//           -> GEMM2(WMMA f16, async global->LDS double-buffered B tiles)
//           -> intra-L2 -> global-L2.

typedef __attribute__((ext_vector_type(16))) _Float16 v16h;
typedef __attribute__((ext_vector_type(8)))  _Float16 v8h;
typedef __attribute__((ext_vector_type(8)))  float    v8f;

namespace {
constexpr int kN = 64, kC = 512, kS = 1200, kSP = 1216, kK = 64;
constexpr float kBnEps = 1e-5f;
constexpr float kE = 2.71828182845904523536f;   // exp(1): softmax of binary spikes
}

__device__ __forceinline__ v16h pack16(v8h lo, v8h hi) {
  v16h r;
#pragma unroll
  for (int i = 0; i < 8; ++i) { r[i] = lo[i]; r[i + 8] = hi[i]; }
  return r;
}

// ---------------------------------------------------------------------------
// Kernel 1: read x[N][C][S] fp32 once; emit
//   x2[N][C][Spad] f16 (same orientation, for GEMM2 B-tiles)
//   xT[N][Spad][C] f16 (transposed,      for GEMM1 A-fragments)
//   inv[N][Spad]   f32 (1/max(||x_s||,1e-12), folded in later)
// Pad region (s in [1200,1216)) written as zeros.  Grid: (19, 64).
// ---------------------------------------------------------------------------
__global__ __launch_bounds__(256) void k_pack(const float* __restrict__ x,
                                              _Float16* __restrict__ x2,
                                              _Float16* __restrict__ xT,
                                              float* __restrict__ inv) {
  const int n = blockIdx.y;
  const int s0 = blockIdx.x * 64;
  const int t = threadIdx.x;
  const int sl = t & 63;     // s within tile (load phase)
  const int cl = t >> 6;     // 0..3
  __shared__ _Float16 lds[16][64 + 8];
  __shared__ float red[4][64];

  const float* xn = x + (size_t)n * kC * kS;
  _Float16* x2n = x2 + (size_t)n * kC * kSP;
  _Float16* xTn = xT + (size_t)n * kSP * kC;
  const int s = s0 + sl;
  const bool sv = (s < kS);
  float ssq = 0.f;

  for (int c0 = 0; c0 < kC; c0 += 16) {
#pragma unroll
    for (int j = 0; j < 4; ++j) {
      const int c = c0 + cl + 4 * j;
      float v = sv ? xn[(size_t)c * kS + s] : 0.f;   // coalesced over s
      ssq += v * v;
      _Float16 h = (_Float16)v;
      x2n[(size_t)c * kSP + s0 + sl] = h;            // coalesced over s
      lds[cl + 4 * j][sl] = h;
    }
    __syncthreads();
    {   // transposed write: 16 c per s, 8B per thread
      const int sr = t >> 2;
      const int co = (t & 3) * 4;
      _Float16* dst = xTn + (size_t)(s0 + sr) * kC + c0 + co;
#pragma unroll
      for (int q = 0; q < 4; ++q) dst[q] = lds[co + q][sr];
    }
    __syncthreads();
  }
  red[cl][sl] = ssq;
  __syncthreads();
  if (t < 64) {
    float ss = red[0][t] + red[1][t] + red[2][t] + red[3][t];
    inv[(size_t)n * kSP + s0 + t] = 1.0f / fmaxf(sqrtf(ss), 1e-12f);
  }
}

// ---------------------------------------------------------------------------
// Kernel 2: GEMM1 (logits = w x) via v_wmma_f32_16x16x32_f16, fused with
// BN + LIF spike + softmax-over-K (binary-spike closed form) to produce
//   a_p[N][K][Spad] f16  (plain soft-assignment, for deterministic asum)
//   a_h[N][K][Spad] f16  (a * inv[s], so GEMM2 absorbs the descriptor norm)
// Block = 64 s x 64 k tile, 8 waves; wave = 16 s-strip x 32 k (2 D-tiles).
// conv_w preloaded to dynamic LDS as f16 (64 KB).
// ---------------------------------------------------------------------------
__global__ __launch_bounds__(256) void k_logits_assign(
    const _Float16* __restrict__ xT, const float* __restrict__ inv,
    const float* __restrict__ wconv, const float* __restrict__ gamma,
    const float* __restrict__ beta, const float* __restrict__ mean,
    const float* __restrict__ var, _Float16* __restrict__ a_h,
    _Float16* __restrict__ a_p) {
  extern __shared__ char smem[];
  _Float16* wl = (_Float16*)smem;                       // [64][512] f16
  float* invl = (float*)(smem + kK * kC * 2);           // [64]
  int* cnt = (int*)(smem + kK * kC * 2 + 64 * 4);       // [64]

  const int n = blockIdx.y;
  const int s0 = blockIdx.x * 64;
  const int t = threadIdx.x;
  const int wv = t >> 5, ln = t & 31;
  const int l16 = ln & 15, hi = ln >> 4;

  for (int i = t; i < kK * kC; i += 256) wl[i] = (_Float16)wconv[i];
  if (t < 64) { invl[t] = inv[(size_t)n * kSP + s0 + t]; cnt[t] = 0; }
  __syncthreads();

  const int ms = (wv >> 1) * 16;      // s-strip within tile
  const int kb = (wv & 1) * 32;       // two k-tiles: kb, kb+16
  const _Float16* aRow = xT + ((size_t)n * kSP + (s0 + ms + l16)) * kC;
  v8f acc[2] = {};

  for (int c0 = 0; c0 < kC; c0 += 32) {
    __builtin_prefetch((const void*)(aRow + c0 + 128), 0, 3);  // near-cache
    // A fragment: 16x32 f16, K = hi*16 + e (matches B)
    v8h a0 = *(const v8h*)(aRow + c0 + hi * 16);
    v8h a1 = *(const v8h*)(aRow + c0 + hi * 16 + 8);
    v16h av = pack16(a0, a1);
    // B fragments from LDS-resident w^T: lane col k, K-rows hi*16+e
    const _Float16* bR0 = &wl[(kb + l16) * kC + c0 + hi * 16];
    const _Float16* bR1 = &wl[(kb + 16 + l16) * kC + c0 + hi * 16];
    v16h bv0 = pack16(*(const v8h*)bR0, *(const v8h*)(bR0 + 8));
    v16h bv1 = pack16(*(const v8h*)bR1, *(const v8h*)(bR1 + 8));
    acc[0] = __builtin_amdgcn_wmma_f32_16x16x32_f16(false, av, false, bv0,
                                                    (short)0, acc[0], false, false);
    acc[1] = __builtin_amdgcn_wmma_f32_16x16x32_f16(false, av, false, bv1,
                                                    (short)0, acc[1], false, false);
  }

  // BN + LIF: spike = H(logit/tau - vth) == (bn(logit) >= 2).  Count spikes
  // per pixel with wave32 ballots -> LDS counters.
#pragma unroll
  for (int j = 0; j < 2; ++j) {
    const int k = kb + 16 * j + l16;
    const float sc = gamma[k] * rsqrtf(var[k] + kBnEps);
    const float bs = beta[k] - mean[k] * sc;
#pragma unroll
    for (int r = 0; r < 8; ++r) {
      const int srow = ms + hi * 8 + r;               // s within block tile
      const float logit = acc[j][r] * invl[srow];     // fold descriptor norm
      const float v = logit * sc + bs;
      const bool spike = (v >= 2.0f);                 // tau*vth
      unsigned m = __builtin_amdgcn_ballot_w32(spike);
      if (ln == 0) {
        atomicAdd(&cnt[ms + r], __popc(m & 0xffffu));       // hi=0 lanes
        atomicAdd(&cnt[ms + 8 + r], __popc(m >> 16));       // hi=1 lanes
      }
      acc[j][r] = spike ? kE : 1.0f;                  // softmax numerator
    }
  }
  __syncthreads();

#pragma unroll
  for (int j = 0; j < 2; ++j) {
    const int k = kb + 16 * j + l16;
    const int sbase = s0 + ms + hi * 8;               // 8-aligned, no straddle
    v8h ap, aw;
#pragma unroll
    for (int r = 0; r < 8; ++r) {
      const int srow = ms + hi * 8 + r;
      const int cc = cnt[srow];
      const float denom = (float)cc * kE + (float)(kK - cc);
      const float aa = acc[j][r] / denom;
      const bool valid = (sbase + r) < kS;
      ap[r] = valid ? (_Float16)aa : (_Float16)0.f;
      aw[r] = valid ? (_Float16)(aa * invl[srow]) : (_Float16)0.f;
    }
    const size_t row = ((size_t)n * kK + k) * kSP + sbase;
    *(v8h*)(a_p + row) = ap;
    *(v8h*)(a_h + row) = aw;
  }
}

// ---------------------------------------------------------------------------
// Kernel 3: asum[n][k] = sum_s a_p  (deterministic tree reduction, no atomics)
// ---------------------------------------------------------------------------
__global__ __launch_bounds__(256) void k_asum(const _Float16* __restrict__ a_p,
                                              float* __restrict__ asum) {
  const int nk = blockIdx.x;
  const int t = threadIdx.x;
  const _Float16* ar = a_p + (size_t)nk * kSP;
  float s = 0.f;
  for (int i = t; i < kS; i += 256) s += (float)ar[i];
  __shared__ float red[256];
  red[t] = s;
  __syncthreads();
  for (int o = 128; o > 0; o >>= 1) { if (t < o) red[t] += red[t + o]; __syncthreads(); }
  if (t == 0) asum[nk] = red[0];
}

// ---------------------------------------------------------------------------
// Kernel 4: GEMM2  vlad[k][c] = sum_s a'[k,s]*x2[c,s] - asum[k]*cent[k,c]
// Block = (n, 128-c chunk); 8 waves; wave = 16 k-strip x 64 c (4 D-tiles).
// B tiles (32 s x 128 c) are staged into LDS with the CDNA5 async DMA path
// (global_load_async_to_lds_b128, ASYNCcnt), double-buffered so the copy of
// tile i+1 overlaps WMMA on tile i.  Each B element hits HBM/L2 once per
// block instead of 4x (one per k-strip wave) and is consumed via ds_load_b128.
// ---------------------------------------------------------------------------
__global__ __launch_bounds__(256) void k_vlad(const _Float16* __restrict__ a_h,
                                              const _Float16* __restrict__ x2,
                                              const float* __restrict__ asum,
                                              const float* __restrict__ cent,
                                              float* __restrict__ vlad) {
  constexpr int kSteps = kSP / 32;                 // 38
  __shared__ _Float16 btile[2][128 * 32];          // 2 x 8KB, [c_local][s_local]
  __shared__ float asl[64];

  const int n = blockIdx.y;
  const int cb = blockIdx.x * 128;
  const int t = threadIdx.x;
  const int wv = t >> 5, ln = t & 31;
  const int l16 = ln & 15, hi = ln >> 4;
  if (t < 64) asl[t] = asum[n * kK + t];

  const int k0 = (wv >> 1) * 16;
  const int chl = (wv & 1) * 64;                   // c-half within block tile
  const _Float16* aRow = a_h + ((size_t)n * kK + k0 + l16) * kSP;
  const _Float16* xB = x2 + (size_t)n * kC * kSP;

  // Async staging: thread t copies 16B of rows (t>>2) and (t>>2)+64.
  const int row0 = t >> 2;                         // 0..63
  const int soff = (t & 3) * 8;                    // half-offset within row
  auto issueTile = [&](int buf, int s0) {
    const _Float16* g0 = xB + (size_t)(cb + row0) * kSP + s0 + soff;
    const _Float16* g1 = xB + (size_t)(cb + 64 + row0) * kSP + s0 + soff;
    unsigned l0 = (unsigned)(size_t)&btile[buf][row0 * 32 + soff];
    unsigned l1 = (unsigned)(size_t)&btile[buf][(64 + row0) * 32 + soff];
    asm volatile("global_load_async_to_lds_b128 %0, %1, off"
                 :: "v"(l0), "v"(g0) : "memory");
    asm volatile("global_load_async_to_lds_b128 %0, %1, off"
                 :: "v"(l1), "v"(g1) : "memory");
  };

  issueTile(0, 0);                                 // prologue: tiles 0 and 1
  issueTile(1, 32);

  v8f acc[4] = {};
  for (int step = 0; step < kSteps; ++step) {
    const int s0 = step * 32;
    if (step + 1 < kSteps) {
      asm volatile("s_wait_asynccnt 0x2" ::: "memory");   // oldest pair done
    } else {
      asm volatile("s_wait_asynccnt 0x0" ::: "memory");
    }
    __syncthreads();                               // tile visible to all waves
    const int buf = step & 1;

    __builtin_prefetch((const void*)(aRow + s0 + 64), 0, 3);
    v8h a0 = *(const v8h*)(aRow + s0 + hi * 16);
    v8h a1 = *(const v8h*)(aRow + s0 + hi * 16 + 8);
    v16h av = pack16(a0, a1);
#pragma unroll
    for (int tt = 0; tt < 4; ++tt) {
      const _Float16* bR = &btile[buf][(chl + 16 * tt + l16) * 32 + hi * 16];
      v16h bv = pack16(*(const v8h*)bR, *(const v8h*)(bR + 8));
      acc[tt] = __builtin_amdgcn_wmma_f32_16x16x32_f16(false, av, false, bv,
                                                       (short)0, acc[tt], false, false);
    }
    __syncthreads();                               // done reading buf
    const int s2 = s0 + 64;
    if (s2 < kSP) issueTile(buf, s2);              // overwrite for step+2
  }

#pragma unroll
  for (int tt = 0; tt < 4; ++tt) {
    const int c = cb + chl + 16 * tt + l16;
#pragma unroll
    for (int r = 0; r < 8; ++r) {
      const int k = k0 + r + 8 * hi;
      const float val = acc[tt][r] - asl[k] * cent[k * kC + c];
      vlad[((size_t)n * kK + k) * kC + c] = val;
    }
  }
}

// ---------------------------------------------------------------------------
// Kernel 5: intra-normalization factor per (n,k) + its contribution to the
// per-n sum of squares (written per-nk; reduced deterministically in k_nss).
// ---------------------------------------------------------------------------
__global__ __launch_bounds__(256) void k_fin1(const float* __restrict__ vlad,
                                              float* __restrict__ r1,
                                              float* __restrict__ ssc) {
  const int nk = blockIdx.x;
  const int t = threadIdx.x;
  const float* vr = vlad + (size_t)nk * kC;
  float ss = vr[t] * vr[t] + vr[t + 256] * vr[t + 256];
  __shared__ float red[256];
  red[t] = ss;
  __syncthreads();
  for (int o = 128; o > 0; o >>= 1) { if (t < o) red[t] += red[t + o]; __syncthreads(); }
  if (t == 0) {
    const float r = 1.0f / fmaxf(sqrtf(red[0]), 1e-12f);
    r1[nk] = r;
    ssc[nk] = red[0] * r * r;
  }
}

__global__ void k_nss(const float* __restrict__ ssc, float* __restrict__ nss) {
  __shared__ float red[64];
  const int n = blockIdx.x, t = threadIdx.x;
  red[t] = ssc[n * kK + t];
  __syncthreads();
  for (int o = 32; o > 0; o >>= 1) { if (t < o) red[t] += red[t + o]; __syncthreads(); }
  if (t == 0) nss[n] = red[0];
}

// Kernel 6: out = vlad * r1[nk] * 1/max(sqrt(nss[n]),1e-12)
__global__ __launch_bounds__(256) void k_fin2(const float* __restrict__ vlad,
                                              const float* __restrict__ r1,
                                              const float* __restrict__ nss,
                                              float* __restrict__ out) {
  const int i = blockIdx.x * 256 + threadIdx.x;
  const int nk = i / kC;
  const int n = nk >> 6;
  const float rn = 1.0f / fmaxf(sqrtf(nss[n]), 1e-12f);
  out[i] = vlad[i] * r1[nk] * rn;
}

// ---------------------------------------------------------------------------
extern "C" void kernel_launch(void* const* d_in, const int* in_sizes, int n_in,
                              void* d_out, int out_size, void* d_ws, size_t ws_size,
                              hipStream_t stream) {
  const float* x = (const float*)d_in[0];        // [64,512,30,40]
  const float* wconv = (const float*)d_in[1];    // [64,512]
  const float* gamma = (const float*)d_in[2];    // [64]
  const float* beta = (const float*)d_in[3];     // [64]
  const float* mean = (const float*)d_in[4];     // [64]
  const float* var = (const float*)d_in[5];      // [64]
  const float* cent = (const float*)d_in[6];     // [64,512]
  float* out = (float*)d_out;                    // [64, 32768]

  char* ws = (char*)d_ws;
  size_t off = 0;
  auto take = [&](size_t bytes) { void* p = ws + off; off += (bytes + 255) & ~(size_t)255; return p; };
  _Float16* x2 = (_Float16*)take((size_t)kN * kC * kSP * 2);   // 79.7 MB
  _Float16* xT = (_Float16*)take((size_t)kN * kSP * kC * 2);   // 79.7 MB
  float* inv = (float*)take((size_t)kN * kSP * 4);
  _Float16* a_h = (_Float16*)take((size_t)kN * kK * kSP * 2);  // 10 MB
  _Float16* a_p = (_Float16*)take((size_t)kN * kK * kSP * 2);  // 10 MB
  float* asum = (float*)take((size_t)kN * kK * 4);
  float* vlad = (float*)take((size_t)kN * kK * kC * 4);        // 8.4 MB
  float* r1 = (float*)take((size_t)kN * kK * 4);
  float* ssc = (float*)take((size_t)kN * kK * 4);
  float* nss = (float*)take((size_t)kN * 4);

  k_pack<<<dim3(19, kN), 256, 0, stream>>>(x, x2, xT, inv);
  const size_t smem1 = (size_t)kK * kC * 2 + 64 * 4 + 64 * 4;  // w + inv + cnt
  k_logits_assign<<<dim3(19, kN), 256, smem1, stream>>>(xT, inv, wconv, gamma,
                                                        beta, mean, var, a_h, a_p);
  k_asum<<<kN * kK, 256, 0, stream>>>(a_p, asum);
  k_vlad<<<dim3(4, kN), 256, 0, stream>>>(a_h, x2, asum, cent, vlad);
  k_fin1<<<kN * kK, 256, 0, stream>>>(vlad, r1, ssc);
  k_nss<<<kN, 64, 0, stream>>>(ssc, nss);
  k_fin2<<<(kN * kK * kC) / 256, 256, 0, stream>>>(vlad, r1, nss, out);
}
